// MultiGraphConv_14766097564100
// MI455X (gfx1250) — matrix operand order repaired and checked by reading, not compiled, not run on hardware
//
#include <hip/hip_runtime.h>
#include <cstdint>
#include <cstddef>

#define B_    16
#define N_    1024
#define F_    64
#define OUT_  128
#define K3_   192
#define EPSF  1e-6f
#define GSTR  200    // padded LDS row stride (bf16) for 192-wide G tile
#define SLABK 128    // K-slab staged in LDS per TDM transfer
#define XSTR  136    // LDS row stride (bf16) for XcT slab: 128 + 8 pad elems

typedef __attribute__((ext_vector_type(16))) __bf16 v16bf;
typedef __attribute__((ext_vector_type(8)))  __bf16 v8bf;
typedef __attribute__((ext_vector_type(8)))  float  v8f;
typedef __attribute__((ext_vector_type(8)))  float  v8fv;
typedef __attribute__((ext_vector_type(4)))  unsigned int v4u;
typedef __attribute__((ext_vector_type(8)))  int v8i;
typedef __attribute__((ext_vector_type(4)))  int v4i;

struct B32 { uint4 lo, hi; };
static_assert(sizeof(B32) == 32, "frag size");

__device__ __forceinline__ unsigned short f2bf(float f) {
  __bf16 h = (__bf16)f;                       // native v_cvt, RNE
  return __builtin_bit_cast(unsigned short, h);
}

__device__ __forceinline__ v16bf fragu(uint4 lo, uint4 hi) {
  B32 t; t.lo = lo; t.hi = hi;
  return __builtin_bit_cast(v16bf, t);
}

// ---------------------------------------------------------------------------
// Kernel 1: per-row degree/diag stats.  One wave32 per adjacency row.
// ---------------------------------------------------------------------------
__global__ void degree_kernel(const float* __restrict__ A,
                              float* __restrict__ inv,
                              float* __restrict__ sv,
                              float* __restrict__ dg) {
  const int lane = threadIdx.x & 31;
  const int w    = threadIdx.x >> 5;
  const int idx  = blockIdx.x * 8 + w;          // row id in [0, B*N)
  const int i    = idx & (N_ - 1);
  const float* row = A + (size_t)idx * N_;

  float sum = 0.f;
#pragma unroll
  for (int it = 0; it < 8; ++it) {
    float4 v = *reinterpret_cast<const float4*>(row + it * 128 + lane * 4);
    sum += v.x + v.y + v.z + v.w;
  }
#pragma unroll
  for (int off = 16; off > 0; off >>= 1) sum += __shfl_xor(sum, off, 32);

  if (lane == 0) {
    float d   = row[i];
    float deg = sum - d + 1.0f;
    inv[idx] = 1.0f / (EPSF + deg);
    sv[idx]  = 1.0f / (EPSF + sqrtf(deg));
    dg[idx]  = d;
  }
}

// ---------------------------------------------------------------------------
// Kernel 2: XcT[b][c][j] = bf16( c<64 ? x[b][j][c] : s[b][j]*x[b][j][c-64] )
// ---------------------------------------------------------------------------
__global__ void xct_kernel(const float* __restrict__ x,
                           const float* __restrict__ sv,
                           unsigned short* __restrict__ XcT) {
  const int b = blockIdx.x >> 7;
  const int c = blockIdx.x & 127;
  const int f = c & 63;
  const bool scaled = c >= 64;
  for (int j = threadIdx.x; j < N_; j += blockDim.x) {
    int ridx = (b << 10) + j;
    float v = x[(size_t)ridx * F_ + f];
    if (scaled) v *= sv[ridx];
    XcT[((size_t)blockIdx.x << 10) + j] = f2bf(v);
  }
}

// ---------------------------------------------------------------------------
// Kernel 3: fused  C = A[b] @ [x | s*x]  (bf16 WMMA; B operand staged in LDS
// via Tensor Data Mover) -> G tile in LDS -> out = relu(G @ W + bias).
// Block: 256 threads = 8 waves, each wave owns a 16x128 row strip.
// ---------------------------------------------------------------------------
__global__ void __launch_bounds__(256)
fused_kernel(const float* __restrict__ A, const float* __restrict__ x,
             const unsigned short* __restrict__ XcT,
             const float* __restrict__ inv, const float* __restrict__ sv,
             const float* __restrict__ dg,
             const float* __restrict__ Wk, const float* __restrict__ bias,
             float* __restrict__ out) {
  __shared__ __align__(16) unsigned short XcS[128 * XSTR]; // TDM slab: 34816 B
  __shared__ __align__(16) unsigned short WT[OUT_ * K3_];  // W^T bf16: 49152 B
  __shared__ __align__(16) unsigned short G[128 * GSTR];   // concat:   51200 B

  const int tid  = threadIdx.x;
  const int lane = tid & 31, w = tid >> 5;
  const int lm   = lane & 15, lh = lane >> 4;
  const int b    = blockIdx.y;
  const int rowBase = blockIdx.x * 128 + w * 16;

  // stage W^T into LDS: WT[o][c]
  for (int idx = tid; idx < K3_ * OUT_; idx += 256) {
    int c = idx >> 7, o = idx & 127;
    WT[o * K3_ + c] = f2bf(Wk[idx]);
  }

  v8f zero8 = {0.f, 0.f, 0.f, 0.f, 0.f, 0.f, 0.f, 0.f};
  v8f acc[8];
#pragma unroll
  for (int t = 0; t < 8; ++t) acc[t] = zero8;

  const float* arow = A + ((size_t)(b * N_ + rowBase + lm)) * N_;
  const unsigned short* xcb = XcT + ((size_t)b << 17);   // XcT[b]: 128x1024 bf16

  // ---- Tensor DMA descriptor (D#), constant parts -------------------------
  // 2D tile: tile_dim0=128 elems (K), tile_dim1=128 rows, data_size=2B,
  // tensor_dim0=1024, stride0=1024; LDS pad: 4 DW after every 64 DW row.
  const unsigned ldsBase = (unsigned)(uintptr_t)(void*)&XcS[0];
  v8i g1 = {0, 0, 0, 0, 0, 0, 0, 0};
  g1[0] = (int)((1u << 16) |            // data_size = 2 bytes
                (1u << 20) |            // pad_enable
                (5u << 22) |            // pad_interval: 64 DWORDs
                (3u << 25));            // pad_amount: 4 DWORDs
  g1[1] = (int)(1024u << 16);           // tensor_dim0[15:0] in [31:16]
  g1[2] = (int)(128u << 16);            // tensor_dim0[31:16]=0 | tensor_dim1=128
  g1[3] = (int)(128u << 16);            // tensor_dim1 hi=0    | tile_dim0=128
  g1[4] = (int)(128u);                  // tile_dim1=128, tile_dim2=0
  g1[5] = (int)(1024u);                 // tensor_dim0_stride = 1024 elems
  v4i gz4 = {0, 0, 0, 0};
  v8i gz8 = {0, 0, 0, 0, 0, 0, 0, 0};

  // ---- GEMM1: K staged in LDS slabs by the TDM ----------------------------
#pragma unroll 1
  for (int s = 0; s < N_ / SLABK; ++s) {
    if (tid < 32) {                         // wave 0 drives the DMA
      unsigned long long gaddr =
          (unsigned long long)(uintptr_t)(xcb + s * SLABK) ;
      v4u g0;
      g0[0] = 1u;                            // count=1, user descriptor
      g0[1] = ldsBase;                       // LDS byte address of slab
      g0[2] = (unsigned)(gaddr & 0xFFFFFFFFu);
      g0[3] = (unsigned)((gaddr >> 32) & 0x1FFFFFFu) | (2u << 30); // type=2
      __builtin_amdgcn_tensor_load_to_lds(g0, g1, gz4, gz4, gz8, 0);
      __builtin_amdgcn_s_wait_tensorcnt(0);
    }
    __syncthreads();                         // slab visible to all waves

    if (s + 2 < N_ / SLABK)
      __builtin_prefetch(arow + (s + 2) * SLABK, 0, 0);

#pragma unroll
    for (int kk = 0; kk < SLABK; kk += 32) {
      // A fragment: 16 f32 -> bf16 via native pk converts
      const float* pa = arow + s * SLABK + kk + lh * 8;
      v8fv f0 = *reinterpret_cast<const v8fv*>(pa);
      v8fv f1 = *reinterpret_cast<const v8fv*>(pa + 16);
      v8bf c0 = __builtin_convertvector(f0, v8bf);
      v8bf c1 = __builtin_convertvector(f1, v8bf);
      v16bf af = __builtin_shufflevector(c0, c1, 0, 1, 2, 3, 4, 5, 6, 7,
                                         8, 9, 10, 11, 12, 13, 14, 15);
      // B fragments from LDS slab, grouped 4-at-a-time for overlap
#pragma unroll
      for (int h = 0; h < 2; ++h) {
        uint4 bl[4], bh[4];
#pragma unroll
        for (int q = 0; q < 4; ++q) {
          int t = h * 4 + q;
          const unsigned short* pb = &XcS[(t * 16 + lm) * XSTR + kk + lh * 8];
          bl[q] = *reinterpret_cast<const uint4*>(pb);
          bh[q] = *reinterpret_cast<const uint4*>(pb + 16);
        }
#pragma unroll
        for (int q = 0; q < 4; ++q) {
          int t = h * 4 + q;
          acc[t] = __builtin_amdgcn_wmma_f32_16x16x32_bf16(
              false, af, false, fragu(bl[q], bh[q]), (short)0, acc[t],
              false, false);
        }
      }
    }
    __syncthreads();                         // done reading slab
  }

  // ---- epilogue 1: form G = [o1 | o2 | o3] in LDS (bf16) ------------------
#pragma unroll
  for (int v = 0; v < 8; ++v) {
    int m = v + lh * 8;
    int r = rowBase + m;
    int ridx = (b << 10) + r;
    float iv = inv[ridx], s1 = sv[ridx], d1 = dg[ridx];
    int rl = w * 16 + m;
#pragma unroll
    for (int t = 0; t < 4; ++t) {
      int f = t * 16 + lm;
      float xv = x[(size_t)ridx * F_ + f];
      float C1 = acc[t][v];
      float C2 = acc[t + 4][v];
      float o1 = C1 - d1 * xv;                       // Az @ x
      float o2 = iv * (o1 + xv);                     // D^-1 A_hat x
      float o3 = s1 * (C2 + s1 * xv * (1.0f - d1));  // D^-1/2 A_hat D^-1/2 x
      G[rl * GSTR + f]       = f2bf(o1);
      G[rl * GSTR + 64 + f]  = f2bf(o2);
      G[rl * GSTR + 128 + f] = f2bf(o3);
    }
  }
  __syncthreads();

  // ---- GEMM2: out = relu(G @ W + bias), operands from LDS -----------------
  v8f acc2[8];
#pragma unroll
  for (int t = 0; t < 8; ++t) acc2[t] = zero8;

#pragma unroll
  for (int k0 = 0; k0 < K3_; k0 += 32) {
    int rl = w * 16 + lm;
    const unsigned short* pg = &G[rl * GSTR + k0 + lh * 8];
    uint4 glo = *reinterpret_cast<const uint4*>(pg);
    uint4 ghi = *reinterpret_cast<const uint4*>(pg + 16);
    v16bf gf = fragu(glo, ghi);
#pragma unroll
    for (int t = 0; t < 8; ++t) {
      const unsigned short* pw = &WT[(t * 16 + lm) * K3_ + k0 + lh * 8];
      uint4 wlo = *reinterpret_cast<const uint4*>(pw);
      uint4 whi = *reinterpret_cast<const uint4*>(pw + 16);
      acc2[t] = __builtin_amdgcn_wmma_f32_16x16x32_bf16(
          false, gf, false, fragu(wlo, whi), (short)0, acc2[t], false, false);
    }
  }

#pragma unroll
  for (int t = 0; t < 8; ++t) {
    int o = t * 16 + lm;
    float bv = bias[o];
#pragma unroll
    for (int v = 0; v < 8; ++v) {
      int r = rowBase + v + lh * 8;
      float val = acc2[t][v] + bv;
      val = val > 0.f ? val : 0.f;
      out[((size_t)(b << 10) + r) * OUT_ + o] = val;
    }
  }
}

// ---------------------------------------------------------------------------
extern "C" void kernel_launch(void* const* d_in, const int* in_sizes, int n_in,
                              void* d_out, int out_size, void* d_ws,
                              size_t ws_size, hipStream_t stream) {
  (void)in_sizes; (void)n_in; (void)out_size; (void)ws_size;
  const float* x    = (const float*)d_in[0];   // [16,1024,64]
  const float* A    = (const float*)d_in[1];   // [16,1024,1024]
  const float* Wk   = (const float*)d_in[2];   // [192,128]
  const float* bias = (const float*)d_in[3];   // [128]
  float* out = (float*)d_out;                  // [16,1024,128]

  // workspace: XcT bf16 (4 MB) + inv/s/diag (3 * 64 KB)
  unsigned short* XcT = (unsigned short*)d_ws;
  char* p = (char*)d_ws + (size_t)B_ * 128 * N_ * sizeof(unsigned short);
  float* inv = (float*)p;  p += (size_t)B_ * N_ * sizeof(float);
  float* sv  = (float*)p;  p += (size_t)B_ * N_ * sizeof(float);
  float* dg  = (float*)p;

  degree_kernel<<<B_ * N_ / 8, 256, 0, stream>>>(A, inv, sv, dg);
  xct_kernel<<<B_ * 128, 256, 0, stream>>>(x, sv, XcT);
  fused_kernel<<<dim3(N_ / 128, B_), 256, 0, stream>>>(
      A, x, XcT, inv, sv, dg, Wk, bias, out);
}
